// QuarotFP16Linear_76407468195936
// MI455X (gfx1250) — compile-verified, matrix operand order checked by reading, not run on hardware
//
#include <hip/hip_runtime.h>

typedef __attribute__((ext_vector_type(16))) _Float16 v16h;
typedef __attribute__((ext_vector_type(8)))  _Float16 v8h;
typedef __attribute__((ext_vector_type(4)))  _Float16 v4h;
typedef __attribute__((ext_vector_type(8)))  float    v8f;
typedef __attribute__((ext_vector_type(4)))  float    v4f;

namespace {
constexpr int T_TOK = 8192;
constexpr int IN_F  = 4096;
constexpr int OUT_F = 11008;
constexpr int NGRP  = 32;       // group size 128 along IN_F

constexpr int BM = 128;         // token tile
constexpr int BN = 128;         // out-feature tile
constexpr int BK = 32;          // K step == WMMA K
constexpr int LDS_K = BK + 8;   // 40 halfs -> 80B row stride (conflict-free frag reads)
}

__global__ __launch_bounds__(256)
void quarot_wmma_gemm(const float* __restrict__ qx,    // [T, I]
                      const float* __restrict__ sx,    // [T, 1]
                      const float* __restrict__ wq,    // [O, I]
                      const float* __restrict__ wscl,  // [O, G]
                      const float* __restrict__ woff,  // [O, G]
                      float* __restrict__ out)         // [T, O]
{
    // Single-buffer staging; 20 KB LDS.
    __shared__ __align__(16) _Float16 lda[BM][LDS_K];
    __shared__ __align__(16) _Float16 ldb[BN][LDS_K];

    const int tid  = threadIdx.x;
    const int lane = tid & 31;
    const int wave = tid >> 5;      // 0..7
    const int wm   = wave >> 2;     // 0..1 : 64-row slice
    const int wn   = wave & 3;      // 0..3 : 32-col slice

    const int bm = blockIdx.y * BM;
    const int bn = blockIdx.x * BN;

    // Loader coordinates: 32 rows per pass, 4 passes; 8 threads x float4 per row.
    const int lrow = tid >> 3;         // 0..31
    const int lcol = (tid & 7) << 2;   // 0,4,...,28

    v8f acc[4][2];
    {
        v8f zero = {};
#pragma unroll
        for (int i = 0; i < 4; ++i)
#pragma unroll
            for (int j = 0; j < 2; ++j) acc[i][j] = zero;
    }

    // Fixed per-thread row bases; K advances via immediate/base-bump offsets.
    const float* qrow = qx + (size_t)(bm + lrow) * IN_F + lcol;
    const float* wrow = wq + (size_t)(bn + lrow) * IN_F + lcol;
    const int    brow = bn + lrow;     // weight-scale row base

    v4f areg[4], breg[4];              // raw fp32 tiles only (scales hoisted per-group)

    auto prefetch = [&](int k0) {
#pragma unroll
        for (int i = 0; i < 4; ++i) {
            areg[i] = *(const v4f*)(qrow + (size_t)(32 * i) * IN_F + k0);
            breg[i] = *(const v4f*)(wrow + (size_t)(32 * i) * IN_F + k0);
        }
    };

    prefetch(0);

    // Fragment coords per ISA 16-bit layout:
    // lane l: row/col = l&15, kb = 8*(l>=16); halves {kb..kb+7} and {kb+16..kb+23}.
    const int lr = lane & 15;
    const int kb = (lane >> 4) << 3;

    for (int g = 0; g < NGRP; ++g) {
        // Group-constant dequant params: loaded once per 4 K-steps.
        float wsr[4], fos[4];
#pragma unroll
        for (int i = 0; i < 4; ++i) {
            const int br = brow + 32 * i;
            const float ws = wscl[br * NGRP + g];
            wsr[i] = ws;
            fos[i] = -woff[br * NGRP + g] * ws;    // (w-off)*ws == w*ws + fos
        }

#pragma unroll
        for (int kk = 0; kk < 4; ++kk) {
            __syncthreads();            // LDS free for refill

            // Dequant (single fp32 FMA) + fp16 pack, stage into LDS.
#pragma unroll
            for (int i = 0; i < 4; ++i) {
                const int row = lrow + 32 * i;
                v4h ah, bh;
#pragma unroll
                for (int c = 0; c < 4; ++c) {
                    ah[c] = (_Float16)areg[i][c];                      // sx folded at epilogue
                    bh[c] = (_Float16)(breg[i][c] * wsr[i] + fos[i]);
                }
                *(v4h*)&lda[row][lcol] = ah;
                *(v4h*)&ldb[row][lcol] = bh;
            }
            __syncthreads();

            // Issue next tile's loads early; consumed only at next stage.
            if (!(g == NGRP - 1 && kk == 3))
                prefetch(g * 128 + kk * 32 + 32);

            v16h af[4];
#pragma unroll
            for (int i = 0; i < 4; ++i) {
                const _Float16* p = &lda[wm * 64 + i * 16 + lr][0];
                union { v16h v; v8h h[2]; } u;
                u.h[0] = *(const v8h*)(p + kb);
                u.h[1] = *(const v8h*)(p + kb + 16);
                af[i] = u.v;
            }
            v16h bf[2];
#pragma unroll
            for (int j = 0; j < 2; ++j) {
                const _Float16* p = &ldb[wn * 32 + j * 16 + lr][0];
                union { v16h v; v8h h[2]; } u;
                u.h[0] = *(const v8h*)(p + kb);
                u.h[1] = *(const v8h*)(p + kb + 16);
                bf[j] = u.v;
            }

#pragma unroll
            for (int i = 0; i < 4; ++i)
#pragma unroll
                for (int j = 0; j < 2; ++j)
                    acc[i][j] = __builtin_amdgcn_wmma_f32_16x16x32_f16(
                        /*neg_a=*/false, af[i], /*neg_b=*/false, bf[j],
                        /*c_mod=*/(short)0, acc[i][j],
                        /*reuse_a=*/false, /*reuse_b=*/false);
        }
    }

    // Epilogue: D layout -> lane l: N = l&15, M = vgpr + 8*(l>=16).
    // Per-token scale folded here (WMMA linear in A): 64 muls/lane total.
    const int mh = (lane >> 4) << 3;
#pragma unroll
    for (int i = 0; i < 4; ++i) {
        const int mbase = bm + wm * 64 + i * 16 + mh;
        float sxv[8];
#pragma unroll
        for (int r = 0; r < 8; ++r) sxv[r] = sx[mbase + r];
#pragma unroll
        for (int j = 0; j < 2; ++j) {
            const int n = bn + wn * 32 + j * 16 + lr;
#pragma unroll
            for (int r = 0; r < 8; ++r)
                out[(size_t)(mbase + r) * OUT_F + n] = acc[i][j][r] * sxv[r];
        }
    }
}

extern "C" void kernel_launch(void* const* d_in, const int* in_sizes, int n_in,
                              void* d_out, int out_size, void* d_ws, size_t ws_size,
                              hipStream_t stream) {
    (void)in_sizes; (void)n_in; (void)out_size; (void)d_ws; (void)ws_size;
    const float* qx   = (const float*)d_in[0];
    const float* sx   = (const float*)d_in[1];
    const float* wq   = (const float*)d_in[2];
    const float* wscl = (const float*)d_in[3];
    const float* woff = (const float*)d_in[4];
    float* out = (float*)d_out;

    dim3 grid(OUT_F / BN, T_TOK / BM);   // (86, 64)
    quarot_wmma_gemm<<<grid, 256, 0, stream>>>(qx, sx, wq, wscl, woff, out);
}
